// LabelAttention_37752762532267
// MI455X (gfx1250) — matrix operand order repaired and checked by reading, not compiled, not run on hardware
//
#include <hip/hip_runtime.h>

// ---- problem constants ----
#define B_  4
#define S_  3072
#define D_  768
#define H_  4
#define E_  2000
#define EH_ (E_*H_)   // 8000
#define BS_ (B_*S_)   // 12288
#define ROWS_ 32      // eh-rows per workgroup in fused kernel (8 labels)

typedef __attribute__((ext_vector_type(16))) __bf16 v16bf;
typedef __attribute__((ext_vector_type(8)))  __bf16 v8bf;
typedef __attribute__((ext_vector_type(8)))  float  v8f;

// Build a 16-element bf16 WMMA fragment from two contiguous 8-element (16B) runs.
__device__ __forceinline__ v16bf load_frag_2x8(const __bf16* p0, const __bf16* p1) {
  v8bf lo = *(const v8bf*)p0;
  v8bf hi = *(const v8bf*)p1;
  v16bf f;
#pragma unroll
  for (int i = 0; i < 8; ++i) { f[i] = lo[i]; f[i + 8] = hi[i]; }
  return f;
}

// ---------------- K0a: fp32 -> bf16 convert ----------------
__global__ void cvt_f32_bf16(const float* __restrict__ src, __bf16* __restrict__ dst, int n) {
  int i = blockIdx.x * blockDim.x + threadIdx.x;
  int stride = gridDim.x * blockDim.x;
  for (; i < n; i += stride) dst[i] = (__bf16)src[i];
}

// ---------------- K0b: hT[b,d,s] = bf16(h[b,s,d]) ----------------
__global__ void transpose_h(const float* __restrict__ h, __bf16* __restrict__ hT) {
  __shared__ float tile[16][17];
  int b  = blockIdx.z;
  int d0 = blockIdx.x * 16;
  int s0 = blockIdx.y * 16;
  int tx = threadIdx.x, ty = threadIdx.y;
  tile[ty][tx] = h[((long)(b * S_ + s0 + ty)) * D_ + d0 + tx];  // coalesced over d
  __syncthreads();
  hT[((long)(b * D_ + d0 + ty)) * S_ + s0 + tx] = (__bf16)tile[tx][ty];  // coalesced over s
}

// ---------------- K1: z = tanh(h @ W^T + b), bf16 out ----------------
__global__ __launch_bounds__(256) void z_gemm(const __bf16* __restrict__ hb,
                                              const __bf16* __restrict__ Wb,
                                              const float*  __restrict__ bias,
                                              __bf16* __restrict__ zb) {
  int wave = threadIdx.x >> 5;
  int lane = threadIdx.x & 31;
  int hf = lane >> 4, ln = lane & 15;
  int tile = blockIdx.x * 8 + wave;              // 768*48 tiles total
  int nt = tile % (D_ / 16);
  int mt = tile / (D_ / 16);

  const __bf16* arow = hb + (long)(mt * 16 + ln) * D_;   // A: rows of h
  const __bf16* brow = Wb + (long)(nt * 16 + ln) * D_;   // B[k,n]=W[n,k]: rows of W
  v8f c = {};
  for (int k0 = 0; k0 < D_; k0 += 32) {
    v16bf a = load_frag_2x8(arow + k0 + hf * 8,  arow + k0 + 16 + hf * 8);
    v16bf b = load_frag_2x8(brow + k0 + hf * 8, brow + k0 + 16 + hf * 8);
    c = __builtin_amdgcn_wmma_f32_16x16x32_bf16(false, a, false, b, (short)0, c, false, false);
  }
  float bv = bias[nt * 16 + ln];
#pragma unroll
  for (int r = 0; r < 8; ++r) {
    float v = tanhf(c[r] + bv);
    zb[(long)(mt * 16 + r + hf * 8) * D_ + nt * 16 + ln] = (__bf16)v;
  }
}

// ---------------- K2: fused score GEMM + softmax + alpha@h + head max-pool ----------------
// One workgroup = (batch b, 32 (e,h) rows = 8 labels). Scores live only in LDS.
// z_b and hT_b are each streamed exactly once per workgroup (halved L2 traffic vs 16-row tiles).
__global__ __launch_bounds__(256) void label_attn(const __bf16* __restrict__ ub,
                                                  const __bf16* __restrict__ zb,
                                                  const __bf16* __restrict__ hT,
                                                  const int*    __restrict__ amask,
                                                  float* __restrict__ mOut) {
  __shared__ __bf16 sU[ROWS_ * D_];     // 48 KB: label_feat tile
  __shared__ __bf16 sS[ROWS_ * S_];     // 192 KB: scores -> alpha (in place)
  __shared__ float  sRed[ROWS_][8];
  __shared__ float  sStat[ROWS_][2];    // [row]{max,sum}

  int blk = blockIdx.x;                 // 0..999
  int b   = blk / (EH_ / ROWS_);        // 250 eh-tiles per batch
  int et  = blk % (EH_ / ROWS_);
  int eh0 = et * ROWS_;
  int tid = threadIdx.x;
  int wave = tid >> 5, lane = tid & 31, hf = lane >> 4, ln = lane & 15;

  // stage U tile -> LDS (contiguous 49152 B)
  {
    const uint4* src = (const uint4*)(ub + (long)eh0 * D_);
    uint4* dst = (uint4*)sU;
    for (int i = tid; i < (ROWS_ * D_) / 8; i += 256) dst[i] = src[i];
  }
  __syncthreads();

  // ---- Phase 1: scores[32, S] = U_tile @ z_b^T (masked), bf16 into LDS ----
  // Each wave owns an s-tile; computes both 16-row M-halves off one B fragment.
  for (int st = wave; st < S_ / 16; st += 8) {
    const __bf16* brow = zb + (long)(b * S_ + st * 16 + ln) * D_;
    v8f c0 = {}, c1 = {};
    for (int k0 = 0; k0 < D_; k0 += 32) {
      v16bf bb = load_frag_2x8(brow + k0 + hf * 8, brow + k0 + 16 + hf * 8);
      v16bf a0 = load_frag_2x8(sU + ln * D_ + k0 + hf * 8,
                               sU + ln * D_ + k0 + 16 + hf * 8);
      v16bf a1 = load_frag_2x8(sU + (16 + ln) * D_ + k0 + hf * 8,
                               sU + (16 + ln) * D_ + k0 + 16 + hf * 8);
      c0 = __builtin_amdgcn_wmma_f32_16x16x32_bf16(false, a0, false, bb, (short)0, c0, false, false);
      c1 = __builtin_amdgcn_wmma_f32_16x16x32_bf16(false, a1, false, bb, (short)0, c1, false, false);
    }
    int s = st * 16 + ln;
    int mk = amask[b * S_ + s];
#pragma unroll
    for (int r = 0; r < 8; ++r) {
      float v0 = mk ? c0[r] : -1.0e6f;
      float v1 = mk ? c1[r] : -1.0e6f;
      sS[(r + hf * 8) * S_ + s]      = (__bf16)v0;
      sS[(16 + r + hf * 8) * S_ + s] = (__bf16)v1;
    }
  }
  __syncthreads();

  // ---- Phase 2: softmax over S per row (8 threads/row, fp32 stats) ----
  int row = tid >> 3, col = tid & 7;
  float mx = -3.0e38f;
  for (int s = col; s < S_; s += 8) mx = fmaxf(mx, (float)sS[row * S_ + s]);
  sRed[row][col] = mx;
  __syncthreads();
  if (tid < ROWS_) {
    float m2 = sRed[tid][0];
    for (int i = 1; i < 8; ++i) m2 = fmaxf(m2, sRed[tid][i]);
    sStat[tid][0] = m2;
  }
  __syncthreads();
  float rowmax = sStat[row][0];
  float sm = 0.f;
  for (int s = col; s < S_; s += 8) sm += expf((float)sS[row * S_ + s] - rowmax);
  sRed[row][col] = sm;
  __syncthreads();
  if (tid < ROWS_) {
    float t = 0.f;
    for (int i = 0; i < 8; ++i) t += sRed[tid][i];
    sStat[tid][1] = fmaxf(t, 1e-20f);
  }
  __syncthreads();
  float inv = 1.0f / sStat[row][1];
  rowmax = sStat[row][0];
  for (int s = col; s < S_; s += 8) {
    float a = expf((float)sS[row * S_ + s] - rowmax) * inv;
    sS[row * S_ + s] = (__bf16)a;   // alpha, in place
  }
  __syncthreads();

  // ---- Phase 3: M[32, 768] = alpha[32, S] @ h_b[S, 768] ----
  // Each wave owns 6 n-tiles; accumulates both M-halves off one B fragment.
  v8f zero = {};
  v8f acc0[6] = {zero, zero, zero, zero, zero, zero};
  v8f acc1[6] = {zero, zero, zero, zero, zero, zero};
  for (int k0 = 0; k0 < S_; k0 += 32) {
    v16bf a0 = load_frag_2x8(sS + ln * S_ + k0 + hf * 8,
                             sS + ln * S_ + k0 + 16 + hf * 8);
    v16bf a1 = load_frag_2x8(sS + (16 + ln) * S_ + k0 + hf * 8,
                             sS + (16 + ln) * S_ + k0 + 16 + hf * 8);
#pragma unroll
    for (int t = 0; t < 6; ++t) {
      int nt = wave + t * 8;
      const __bf16* brow = hT + (long)(b * D_ + nt * 16 + ln) * S_;  // B[k=s,n=d] from hT rows
      v16bf bb = load_frag_2x8(brow + k0 + hf * 8, brow + k0 + 16 + hf * 8);
      acc0[t] = __builtin_amdgcn_wmma_f32_16x16x32_bf16(false, a0, false, bb, (short)0, acc0[t], false, false);
      acc1[t] = __builtin_amdgcn_wmma_f32_16x16x32_bf16(false, a1, false, bb, (short)0, acc1[t], false, false);
    }
  }

  // ---- Phase 4: max over 4 heads (4 consecutive rows) directly from C layout ----
  // In each 16-row C tile: hf=0 lanes hold rows 0..7 (2 labels), hf=1 lanes rows 8..15.
#pragma unroll
  for (int t = 0; t < 6; ++t) {
    int nt = wave + t * 8;
    int d  = nt * 16 + ln;
    // M-half 0: labels et*8 + hf*2 + {0,1}
    {
      float m0 = fmaxf(fmaxf(acc0[t][0], acc0[t][1]), fmaxf(acc0[t][2], acc0[t][3]));
      float m1 = fmaxf(fmaxf(acc0[t][4], acc0[t][5]), fmaxf(acc0[t][6], acc0[t][7]));
      int e0 = et * 8 + hf * 2;
      mOut[((long)(b * E_ + e0)) * D_ + d]     = m0;
      mOut[((long)(b * E_ + e0 + 1)) * D_ + d] = m1;
    }
    // M-half 1: labels et*8 + 4 + hf*2 + {0,1}
    {
      float m0 = fmaxf(fmaxf(acc1[t][0], acc1[t][1]), fmaxf(acc1[t][2], acc1[t][3]));
      float m1 = fmaxf(fmaxf(acc1[t][4], acc1[t][5]), fmaxf(acc1[t][6], acc1[t][7]));
      int e0 = et * 8 + 4 + hf * 2;
      mOut[((long)(b * E_ + e0)) * D_ + d]     = m0;
      mOut[((long)(b * E_ + e0 + 1)) * D_ + d] = m1;
    }
  }
}

// ---------------- K3: logits[b,e] = dot(cls_w[e], m[b,e]) + cls_b[e] ----------------
__global__ __launch_bounds__(256) void logits_kernel(const float* __restrict__ mIn,
                                                     const float* __restrict__ cw,
                                                     const float* __restrict__ cb,
                                                     float* __restrict__ out) {
  __shared__ float red[256];
  int idx = blockIdx.x;            // b*E_ + e
  int e = idx % E_;
  const float* mrow = mIn + (long)idx * D_;
  const float* wrow = cw + (long)e * D_;
  float s = 0.f;
  for (int d = threadIdx.x; d < D_; d += 256) s += mrow[d] * wrow[d];
  red[threadIdx.x] = s;
  __syncthreads();
  for (int off = 128; off > 0; off >>= 1) {
    if (threadIdx.x < off) red[threadIdx.x] += red[threadIdx.x + off];
    __syncthreads();
  }
  if (threadIdx.x == 0) out[idx] = red[0] + cb[e];
}

// ---------------- workspace layout (bytes) ----------------
#define WS_UB   0                                   // 8000*768*2  = 12,288,000
#define WS_WB   (WS_UB + (size_t)EH_ * D_ * 2)      // 768*768*2   =  1,179,648
#define WS_HB   (WS_WB + (size_t)D_ * D_ * 2)       // 12288*768*2 = 18,874,368
#define WS_HT   (WS_HB + (size_t)BS_ * D_ * 2)      // 4*768*3072*2= 18,874,368
#define WS_ZB   (WS_HT + (size_t)B_ * D_ * S_ * 2)  // 12288*768*2 = 18,874,368

extern "C" void kernel_launch(void* const* d_in, const int* in_sizes, int n_in,
                              void* d_out, int out_size, void* d_ws, size_t ws_size,
                              hipStream_t stream) {
  const float* h    = (const float*)d_in[0];
  const int*   am   = (const int*)  d_in[1];
  const float* lf   = (const float*)d_in[2];
  const float* Ww   = (const float*)d_in[3];
  const float* Wbia = (const float*)d_in[4];
  const float* clsw = (const float*)d_in[5];
  const float* clsb = (const float*)d_in[6];
  float* out = (float*)d_out;           // [logits 8000][m 6,144,000]

  char* ws = (char*)d_ws;
  __bf16* ub = (__bf16*)(ws + WS_UB);
  __bf16* Wb = (__bf16*)(ws + WS_WB);
  __bf16* hb = (__bf16*)(ws + WS_HB);
  __bf16* hT = (__bf16*)(ws + WS_HT);
  __bf16* zb = (__bf16*)(ws + WS_ZB);

  // K0: converts + transpose
  cvt_f32_bf16<<<2048, 256, 0, stream>>>(lf, ub, EH_ * D_);
  cvt_f32_bf16<<<1024, 256, 0, stream>>>(Ww, Wb, D_ * D_);
  cvt_f32_bf16<<<2048, 256, 0, stream>>>(h, hb, BS_ * D_);
  transpose_h<<<dim3(D_ / 16, S_ / 16, B_), dim3(16, 16), 0, stream>>>(h, hT);

  // K1: z = tanh(h @ W^T + b)
  z_gemm<<<(BS_ / 16) * (D_ / 16) / 8, 256, 0, stream>>>(hb, Wb, Wbia, zb);

  // K2: fused label attention -> m (at out+8000)
  label_attn<<<B_ * (EH_ / ROWS_), 256, 0, stream>>>(ub, zb, hT, am, out + 8000);

  // K3: logits
  logits_kernel<<<B_ * E_, 256, 0, stream>>>(out + 8000, clsw, clsb, out);
}
